// GraphMultiHeadAttentionLayer_39779987096384
// MI455X (gfx1250) — matrix operand order repaired and checked by reading, not compiled, not run on hardware
//
#include <hip/hip_runtime.h>
#include <hip/hip_bf16.h>

typedef __attribute__((ext_vector_type(16))) _Float16 v16h;
typedef __attribute__((ext_vector_type(8)))  _Float16 v8h;
typedef __attribute__((ext_vector_type(4)))  _Float16 v4h;
typedef __attribute__((ext_vector_type(8)))  float    v8f;
typedef __attribute__((ext_vector_type(4)))  float    v4f;

#define B_    16
#define N_    1024
#define DIN_  128
#define H_    4
#define D_    32
#define HD_   128
#define JC_   256          // j-chunk size in k_attn

// ---------------------------------------------------------------------------
// Kernel 1: h = x @ W + b, stored TRANSPOSED as f16: hT[b, col, n]
// 8 waves / block, each wave owns a 16-row tile, all 128 output cols.
// ---------------------------------------------------------------------------
__global__ __launch_bounds__(256) void k_gemm_h(
    const float* __restrict__ x,     // [B,N,DIN]
    const float* __restrict__ W,     // [DIN,HD]
    const float* __restrict__ bias,  // [HD]
    _Float16* __restrict__ hT)       // [B,HD,N]
{
  __shared__ __align__(32) _Float16 WT[DIN_ * HD_];  // 32KB, WT[col*128+k]
  const int t = threadIdx.x;
  for (int idx = t; idx < DIN_ * HD_; idx += 256) {
    int k = idx / HD_, col = idx % HD_;
    WT[col * DIN_ + k] = (_Float16)W[idx];
  }
  __syncthreads();

  const int wave = t >> 5;
  const int lane = t & 31;
  const int hi   = lane >> 4;
  const int lm   = lane & 15;
  const int rowtile = blockIdx.x * 8 + wave;        // 0..1023
  const long row0   = (long)rowtile * 16;
  const float* xrow = x + (row0 + lm) * DIN_;       // A row = lane%16

  v8f acc[8] = {};
  for (int ks = 0; ks < DIN_; ks += 32) {
    const int g0 = ks + hi * 8;                     // A: K={g0..g0+7, +16}
    v4f x0 = *(const v4f*)(xrow + g0);
    v4f x1 = *(const v4f*)(xrow + g0 + 4);
    v4f x2 = *(const v4f*)(xrow + g0 + 16);
    v4f x3 = *(const v4f*)(xrow + g0 + 20);
    v16h a;
#pragma unroll
    for (int e = 0; e < 4; ++e) {
      a[e]      = (_Float16)x0[e];
      a[4 + e]  = (_Float16)x1[e];
      a[8 + e]  = (_Float16)x2[e];
      a[12 + e] = (_Float16)x3[e];
    }
#pragma unroll
    for (int ct = 0; ct < 8; ++ct) {
      const int col = ct * 16 + lm;
      v8h blo = *(const v8h*)&WT[col * DIN_ + g0];
      v8h bhi = *(const v8h*)&WT[col * DIN_ + g0 + 16];
      v16h bf = __builtin_shufflevector(blo, bhi,
          0, 1, 2, 3, 4, 5, 6, 7, 8, 9, 10, 11, 12, 13, 14, 15);
      acc[ct] = __builtin_amdgcn_wmma_f32_16x16x32_f16(
          false, a, false, bf, (short)0, acc[ct], false, false);
    }
  }

  const int b  = (int)(row0 / N_);
  const int n0 = (int)(row0 % N_) + hi * 8;
#pragma unroll
  for (int ct = 0; ct < 8; ++ct) {
    const int col = ct * 16 + lm;
    const float bb = bias[col];
    v8h o;
#pragma unroll
    for (int r = 0; r < 8; ++r) o[r] = (_Float16)(acc[ct][r] + bb);
    *(v8h*)&hT[((long)b * HD_ + col) * N_ + n0] = o;
  }
}

// ---------------------------------------------------------------------------
// Kernel 1b: e_i[b,h,n] = sum_d hT[b,h*32+d,n]*a[d];  e_j with a[D+d]
// ---------------------------------------------------------------------------
__global__ __launch_bounds__(256) void k_e(
    const _Float16* __restrict__ hT, const float* __restrict__ a,
    float* __restrict__ ei, float* __restrict__ ej)
{
  const int idx = blockIdx.x * 256 + threadIdx.x;   // (b*H+h)*N + n
  const int n  = idx & (N_ - 1);
  const int bh = idx >> 10;
  const int h  = bh & (H_ - 1);
  const int b  = bh >> 2;
  const _Float16* base = hT + ((long)b * HD_ + h * D_) * N_ + n;
  float s1 = 0.f, s2 = 0.f;
#pragma unroll 4
  for (int d = 0; d < D_; ++d) {
    const float hv = (float)base[(long)d * N_];
    s1 += hv * a[d];
    s2 += hv * a[D_ + d];
  }
  ei[idx] = s1;
  ej[idx] = s2;
}

// ---------------------------------------------------------------------------
// Kernel 1c: hsum[b,col] = sum_n hT[b,col,n]  (uniform-attention fallback)
// ---------------------------------------------------------------------------
__global__ __launch_bounds__(256) void k_hsum(
    const _Float16* __restrict__ hT, float* __restrict__ hsum)
{
  const int idx = blockIdx.x * 256 + threadIdx.x;   // b*HD + col, 2048 total
  const _Float16* r = hT + (long)idx * N_;
  float s = 0.f;
  for (int n = 0; n < N_; n += 16) {
    v16h v = *(const v16h*)&r[n];
#pragma unroll
    for (int e = 0; e < 16; ++e) s += (float)v[e];
  }
  hsum[idx] = s;
}

// ---------------------------------------------------------------------------
// Kernel 2: masked softmax(attn) @ h.
// Block = 128 threads = 4 waves = 4 heads sharing one (b, 16-row) adj bitmask.
// exp computed exactly once per logit; probs staged f16 in LDS per 256-j chunk,
// then consumed by WMMA as two ds_load_b128 per A fragment.
// ---------------------------------------------------------------------------
__global__ __launch_bounds__(128) void k_attn(
    const _Float16* __restrict__ hT,    // [B,HD,N]
    const int*      __restrict__ adj,   // [B,N,N]
    const float*    __restrict__ ei,    // [B,H,N]
    const float*    __restrict__ ej,    // [B,H,N]
    const float*    __restrict__ hsum,  // [B,HD]
    float*          __restrict__ out)   // [B,N,HD]
{
  __shared__ unsigned mask[16 * 32];                    // 2KB   [row][j/32]
  __shared__ __align__(32) _Float16 pbuf[H_][16][JC_];  // 32KB  per-head probs
  __shared__ float srow[H_][16];

  const int bid = blockIdx.x;
  const int b   = bid >> 6;
  const int i0  = (bid & 63) * 16;
  const int t    = threadIdx.x;
  const int wave = t >> 5;              // = head
  const int lane = t & 31;
  const int hi   = lane >> 4;
  const int lm   = lane & 15;

  // --- 1) adj tile -> bitmask in LDS (adj read exactly once) ---------------
  const int* adjbase = adj + ((long)b * N_ + i0) * N_;
  for (int c = wave; c < 512; c += 4) {
    const int row = c >> 5;
    const int j0  = (c & 31) * 32;
    __builtin_prefetch(adjbase + (long)row * N_ + j0 + 4 * 32, 0, 1);
    const int v = adjbase[(long)row * N_ + j0 + lane];
    const unsigned mb = (unsigned)__ballot(v > 0);      // wave32 low 32 bits
    if (lane == 0) mask[c] = mb;
  }
  __syncthreads();

  const int head = wave;
  const float* eib = ei + ((long)b * H_ + head) * N_ + i0;
  const float* ejb = ej + ((long)b * H_ + head) * N_;
  const _Float16* hT0 = hT + ((long)b * HD_ + head * D_) * N_;

  float eir[16];
#pragma unroll
  for (int r = 0; r < 16; ++r) eir[r] = eib[r];

  float sums[16];
#pragma unroll
  for (int r = 0; r < 16; ++r) sums[r] = 0.f;

  v8f c0 = {}, c1 = {};
  const int bitb = (lane & 7) * 4;      // bit base within mask dword

  for (int jc = 0; jc < N_; jc += JC_) {
    // --- 2a) build unnormalized probs for this chunk into LDS --------------
    // per (row,g): lane covers j = jc + g*128 + lane*4 .. +3
    v4f ejv[2];
#pragma unroll
    for (int g = 0; g < 2; ++g)
      ejv[g] = *(const v4f*)&ejb[jc + g * 128 + lane * 4];

#pragma unroll
    for (int row = 0; row < 16; ++row) {
      const float e_i = eir[row];
#pragma unroll
      for (int g = 0; g < 2; ++g) {
        const int jo = g * 128 + lane * 4;            // offset within chunk
        const unsigned mb = mask[row * 32 + ((jc + jo) >> 5)];
        v4h pv;
        float sacc = 0.f;
#pragma unroll
        for (int e = 0; e < 4; ++e) {
          float v = e_i + ejv[g][e];
          v = v > 0.f ? v : 0.2f * v;                  // leaky relu
          float p = ((mb >> (bitb + e)) & 1u) ? __expf(v) : 0.f;
          sacc += p;
          pv[e] = (_Float16)p;
        }
        sums[row] += sacc;
        *(v4h*)&pbuf[wave][row][jo] = pv;
      }
    }

    // --- 2b) WMMA over the chunk: C += P_chunk @ h_chunk -------------------
    const int kb = hi * 8;                             // A K-group base
#pragma unroll
    for (int ks = 0; ks < JC_; ks += 32) {
      v8h alo = *(const v8h*)&pbuf[wave][lm][ks + kb];        // K kb..kb+7
      v8h ahi = *(const v8h*)&pbuf[wave][lm][ks + kb + 16];   // K kb+16..+23
      v16h a = __builtin_shufflevector(alo, ahi,
          0, 1, 2, 3, 4, 5, 6, 7, 8, 9, 10, 11, 12, 13, 14, 15);
      const _Float16* p0 = &hT0[(long)lm * N_ + jc + ks + kb];
      const _Float16* p1 = &hT0[(long)(16 + lm) * N_ + jc + ks + kb];
      v16h b0 = __builtin_shufflevector(*(const v8h*)p0, *(const v8h*)(p0 + 16),
          0, 1, 2, 3, 4, 5, 6, 7, 8, 9, 10, 11, 12, 13, 14, 15);
      v16h b1 = __builtin_shufflevector(*(const v8h*)p1, *(const v8h*)(p1 + 16),
          0, 1, 2, 3, 4, 5, 6, 7, 8, 9, 10, 11, 12, 13, 14, 15);
      c0 = __builtin_amdgcn_wmma_f32_16x16x32_f16(false, a, false, b0, (short)0, c0, false, false);
      c1 = __builtin_amdgcn_wmma_f32_16x16x32_f16(false, a, false, b1, (short)0, c1, false, false);
    }
  }

  // --- 3) reduce row sums across the wave ---------------------------------
#pragma unroll
  for (int row = 0; row < 16; ++row) {
    float s = sums[row];
    for (int o = 16; o > 0; o >>= 1) s += __shfl_xor(s, o, 32);
    if (lane == 0) srow[wave][row] = s;
  }
  __syncthreads();

  // --- 4) normalize and store; all-masked row -> uniform 1/N fallback ------
  const float hs0 = hsum[b * HD_ + head * D_ + lm];
  const float hs1 = hsum[b * HD_ + head * D_ + 16 + lm];
  float* ob = out + ((long)b * N_ + i0) * HD_ + head * D_;
#pragma unroll
  for (int r = 0; r < 8; ++r) {
    const int rr = r + hi * 8;                         // C row = vgpr + 8*half
    const float s = srow[wave][rr];
    const float inv = (s > 0.f) ? (1.0f / s) : 0.f;
    const float u   = (s > 0.f) ? 0.f : (1.0f / (float)N_);
    ob[(long)rr * HD_ + lm]      = c0[r] * inv + hs0 * u;
    ob[(long)rr * HD_ + 16 + lm] = c1[r] * inv + hs1 * u;
  }
}

// ---------------------------------------------------------------------------
extern "C" void kernel_launch(void* const* d_in, const int* in_sizes, int n_in,
                              void* d_out, int out_size, void* d_ws, size_t ws_size,
                              hipStream_t stream) {
  const float* x    = (const float*)d_in[0];
  const int*   adj  = (const int*)  d_in[1];
  const float* W    = (const float*)d_in[2];
  const float* bias = (const float*)d_in[3];
  const float* a    = (const float*)d_in[4];
  float* out        = (float*)d_out;

  // ws: hT f16 (4MB) | ei f32 (256KB) | ej f32 (256KB) | hsum f32 (8KB)
  _Float16* hT = (_Float16*)d_ws;
  float* ei   = (float*)((char*)d_ws + ((size_t)B_ * HD_ * N_ * sizeof(_Float16)));
  float* ej   = ei + (size_t)B_ * H_ * N_;
  float* hsum = ej + (size_t)B_ * H_ * N_;

  k_gemm_h<<<(B_ * N_ / 16) / 8, 256, 0, stream>>>(x, W, bias, hT);
  k_e<<<(B_ * H_ * N_) / 256, 256, 0, stream>>>(hT, a, ei, ej);
  k_hsum<<<(B_ * HD_) / 256, 256, 0, stream>>>(hT, hsum);
  k_attn<<<B_ * (N_ / 16), 128, 0, stream>>>(hT, adj, ei, ej, hsum, out);
}
